// VectorQuantizerEMA_5789615915724
// MI455X (gfx1250) — compile-verified, matrix operand order checked by reading, not run on hardware
//
#include <hip/hip_runtime.h>

typedef __attribute__((ext_vector_type(16))) __bf16 v16bf;
typedef __attribute__((ext_vector_type(8)))  float  v8f;

#define NCODES 512
#define CDIM   64
#define HW     4096        // H*W
#define IMG    262144      // C*H*W per batch image
#define NELEM  4194304.0f  // B*C*H*W

// ---------------------------------------------------------------------------
// Prep: fold -2 into codebook, split to bf16 hi/lo, compute ||e||^2, zero loss.
// ---------------------------------------------------------------------------
__global__ void vq_prep(const float* __restrict__ emb,
                        __bf16* __restrict__ ehi, __bf16* __restrict__ elo,
                        float* __restrict__ enorm2, float* __restrict__ out) {
  int k = blockIdx.x * blockDim.x + threadIdx.x;
  if (k == 0) out[0] = 0.0f;                 // loss accumulator
  if (k >= NCODES) return;
  const float* row = emb + k * CDIM;
  float s = 0.0f;
  for (int c = 0; c < CDIM; ++c) {
    float e = row[c];
    s += e * e;
    float t = -2.0f * e;
    __bf16 h = (__bf16)t;
    __bf16 l = (__bf16)(t - (float)h);
    ehi[k * CDIM + c] = h;
    elo[k * CDIM + c] = l;
  }
  enorm2[k] = s;
}

// Async DMA of 16 bytes/lane from global to LDS (ASYNCcnt-tracked).
__device__ __forceinline__ void async_b128(unsigned lds_off, const void* gaddr) {
  asm volatile("global_load_async_to_lds_b128 %0, %1, off"
               :: "v"(lds_off), "v"(gaddr) : "memory");
}

// ---------------------------------------------------------------------------
// Main: fused distance GEMM (bf16 WMMA, hi/lo split) + argmin + gather + loss.
// One wave32 per 16-pixel tile; 8 waves/block share async-staged B tiles.
// ---------------------------------------------------------------------------
__launch_bounds__(256)
__global__ void vq_main(const float* __restrict__ x,
                        const float* __restrict__ emb,
                        const __bf16* __restrict__ ehi,
                        const __bf16* __restrict__ elo,
                        const float* __restrict__ enorm2,
                        float* __restrict__ out) {
  // Double-buffered B staging: [buf][hi 2KB | lo 2KB]
  __shared__ __align__(16) char smem[2][4096];

  const int lane = threadIdx.x & 31;
  const int wave = threadIdx.x >> 5;
  const int tile = blockIdx.x * 8 + wave;    // 0..4095
  const int P    = tile * 16;                // base flat pixel (b,h,w)
  const int b    = P >> 12;
  const int hw0  = P & 4095;                 // 16-aligned, never straddles a row
  const float* xbase = x + (size_t)b * IMG + hw0;   // + c*HW + p

  const unsigned lbase0 = (unsigned)(uintptr_t)&smem[0][0];
  const unsigned lbase1 = (unsigned)(uintptr_t)&smem[1][0];

  // ---- Build A operand (16 pixels x 64 channels), bf16 hi/lo split ----
  // ISA 16-bit A 16x32 layout: lanes 0-15: M=lane, K in {kg..kg+7, kg+16..kg+23}
  const int m  = lane & 15;
  const int kg = (lane >> 4) * 8;            // 0 or 8
  v16bf ahi[2], alo[2];
#pragma unroll
  for (int q = 0; q < 2; ++q) {              // K chunks of 32
#pragma unroll
    for (int hwd = 0; hwd < 16; ++hwd) {
      int c = 32 * q + kg + hwd + ((hwd >= 8) ? 8 : 0);
      float v = xbase[c * HW + m];           // 16 contiguous pixels per channel
      __bf16 h = (__bf16)v;
      __bf16 l = (__bf16)(v - (float)h);
      ahi[q][hwd] = h;
      alo[q][hwd] = l;
    }
  }

  // ---- Sweep 32 code tiles; running per-lane argmin ----
  float bestS[8];
  int   bestC[8];
#pragma unroll
  for (int j = 0; j < 8; ++j) { bestS[j] = 3.4e38f; bestC[j] = 0; }

  union U { uint4 u[2]; v16bf v; };

  // Prologue: wave 0 stages code-tile 0 into buffer 0.
  if (wave == 0) {
    const char* gh = (const char*)(ehi);
    const char* gl = (const char*)(elo);
#pragma unroll
    for (int i = 0; i < 4; ++i) {
      async_b128(lbase0 + i * 512 + lane * 16,        gh + i * 512 + lane * 16);
      async_b128(lbase0 + 2048 + i * 512 + lane * 16, gl + i * 512 + lane * 16);
    }
  }

  for (int ct = 0; ct < 32; ++ct) {
    const unsigned lcur = (ct & 1) ? lbase1 : lbase0;
    // Stage next tile into the other buffer, then drain current tile's DMA.
    if (wave == 0) {
      if (ct + 1 < 32) {
        const unsigned lnext = (ct & 1) ? lbase0 : lbase1;
        const char* gh = (const char*)(ehi + (ct + 1) * 16 * CDIM);
        const char* gl = (const char*)(elo + (ct + 1) * 16 * CDIM);
#pragma unroll
        for (int i = 0; i < 4; ++i) {
          async_b128(lnext + i * 512 + lane * 16,        gh + i * 512 + lane * 16);
          async_b128(lnext + 2048 + i * 512 + lane * 16, gl + i * 512 + lane * 16);
        }
        asm volatile("s_wait_asynccnt 0x8" ::: "memory");  // current tile done (in-order)
      } else {
        asm volatile("s_wait_asynccnt 0x0" ::: "memory");
      }
    }
    __syncthreads();                          // B tile visible to all 8 waves

    const __bf16* lp = (const __bf16*)(uintptr_t)0 + 0;   // silence unused warn path
    const __bf16* lhi = (const __bf16*)&smem[ct & 1][0];
    const __bf16* llo = (const __bf16*)&smem[ct & 1][2048];
    (void)lp;

    const int code = ct * 16 + m;            // this lane's column (code index)
    const float en = enorm2[code];
    v8f acc;
#pragma unroll
    for (int j = 0; j < 8; ++j) acc[j] = en; // seed with ||e||^2

#pragma unroll
    for (int q = 0; q < 2; ++q) {
      const __bf16* bh = lhi + m * CDIM + 32 * q + kg;
      const __bf16* bl = llo + m * CDIM + 32 * q + kg;
      U Bh, Bl;
      Bh.u[0] = *(const uint4*)(bh);
      Bh.u[1] = *(const uint4*)(bh + 16);
      Bl.u[0] = *(const uint4*)(bl);
      Bl.u[1] = *(const uint4*)(bl + 16);
      // score += x_hi*(-2e)_hi + x_hi*(-2e)_lo + x_lo*(-2e)_hi  (~f32 accuracy)
      acc = __builtin_amdgcn_wmma_f32_16x16x32_bf16(false, ahi[q], false, Bh.v,
                                                    (short)0, acc, false, false);
      acc = __builtin_amdgcn_wmma_f32_16x16x32_bf16(false, ahi[q], false, Bl.v,
                                                    (short)0, acc, false, false);
      acc = __builtin_amdgcn_wmma_f32_16x16x32_bf16(false, alo[q], false, Bh.v,
                                                    (short)0, acc, false, false);
    }
#pragma unroll
    for (int j = 0; j < 8; ++j) {
      if (acc[j] < bestS[j]) { bestS[j] = acc[j]; bestC[j] = code; }  // first-hit tie-break
    }
    __syncthreads();                          // all waves done before buffer reuse
  }

  // ---- Cross-lane argmin within each 16-lane half (rows 0-7 / 8-15) ----
#pragma unroll
  for (int mask = 8; mask >= 1; mask >>= 1) {
#pragma unroll
    for (int j = 0; j < 8; ++j) {
      float os = __shfl_xor(bestS[j], mask, 32);
      int   oc = __shfl_xor(bestC[j], mask, 32);
      if (os < bestS[j] || (os == bestS[j] && oc < bestC[j])) {
        bestS[j] = os; bestC[j] = oc;
      }
    }
  }

  // Broadcast the 16 winning codes to every lane (rows 0-7 live in lane 0,
  // rows 8-15 in lane 16 after the butterfly).
  int codes[16];
#pragma unroll
  for (int p = 0; p < 16; ++p)
    codes[p] = __shfl(bestC[p & 7], (p < 8) ? 0 : 16, 32);

  // ---- Gather codebook rows, scatter to [B,C,H,W], accumulate loss ----
  float* qout = out + 1;                     // out[0] = loss
  const size_t obase = (size_t)b * IMG + hw0;
  float lsum = 0.0f;
#pragma unroll
  for (int ci = 0; ci < 2; ++ci) {
    const int c = lane + 32 * ci;            // each lane owns 2 channels
    const float* xc = xbase + c * HW;
    float* oc = qout + obase + (size_t)c * HW;
#pragma unroll
    for (int p = 0; p < 16; ++p) {
      float qv = emb[codes[p] * CDIM + c];   // coalesced across lanes
      float d  = qv - xc[p];
      lsum += d * d;
      oc[p] = qv;                            // quantized_st == quantized numerically
    }
  }

  // Wave-wide loss reduction + one atomic per wave.
#pragma unroll
  for (int mask = 16; mask >= 1; mask >>= 1)
    lsum += __shfl_xor(lsum, mask, 32);
  if (lane == 0)
    atomicAdd(out, lsum * (0.25f / NELEM));
}

// ---------------------------------------------------------------------------
extern "C" void kernel_launch(void* const* d_in, const int* in_sizes, int n_in,
                              void* d_out, int out_size, void* d_ws, size_t ws_size,
                              hipStream_t stream) {
  const float* x   = (const float*)d_in[0];   // [16,64,64,64] f32
  const float* emb = (const float*)d_in[1];   // [512,64] f32

  __bf16* ehi    = (__bf16*)d_ws;                                   // 64 KB
  __bf16* elo    = (__bf16*)((char*)d_ws + NCODES * CDIM * 2);      // 64 KB
  float*  enorm2 = (float*)((char*)d_ws + NCODES * CDIM * 4);       // 2 KB
  float*  out    = (float*)d_out;

  vq_prep<<<2, 256, 0, stream>>>(emb, ehi, elo, enorm2, out);
  vq_main<<<512, 256, 0, stream>>>(x, emb, ehi, elo, enorm2, out);
}
